// HSGNNModel_87754771792289
// MI455X (gfx1250) — compile-verified, matrix operand order checked by reading, not run on hardware
//
#include <hip/hip_runtime.h>
#include <hip/hip_bf16.h>
#include <stdint.h>

#define N_NODES 4096
#define N_EDGES 131072
#define IN_C    512
#define HID_C   256
#define OUT_C   128
#define K_SUB   4

typedef __attribute__((ext_vector_type(16))) __bf16 v16bf;
typedef __attribute__((ext_vector_type(8)))  __bf16 v8bf;
typedef __attribute__((ext_vector_type(8)))  float  v8f;
typedef __attribute__((ext_vector_type(4)))  float  v4f;

// ---------------- workspace layout (bytes) ----------------
// deg_inv : N f32
// xbf     : N*IN_C bf16
// w1t     : HID_C*IN_C bf16   (W1 transposed: [n][k])
// hlin    : N*HID_C f32       (x @ W1)
// agg1    : N*HID_C f32       (scatter result + bias)
// hbf     : N*HID_C bf16      (relu(agg1))
// w2t     : OUT_C*HID_C bf16  (W2 transposed)
// h2lin   : N*OUT_C f32       (h @ W2)
static constexpr size_t OFF_DEG  = 0;
static constexpr size_t OFF_XBF  = 64 * 1024;
static constexpr size_t OFF_W1T  = OFF_XBF  + (size_t)N_NODES * IN_C  * 2;
static constexpr size_t OFF_HLIN = OFF_W1T  + (size_t)HID_C   * IN_C  * 2;
static constexpr size_t OFF_AGG1 = OFF_HLIN + (size_t)N_NODES * HID_C * 4;
static constexpr size_t OFF_HBF  = OFF_AGG1 + (size_t)N_NODES * HID_C * 4;
static constexpr size_t OFF_W2T  = OFF_HBF  + (size_t)N_NODES * HID_C * 2;
static constexpr size_t OFF_H2   = OFF_W2T  + (size_t)OUT_C   * HID_C * 2;

// ---------------- degree kernels ----------------
__global__ void k_init_deg(float* __restrict__ deg) {
    int n = blockIdx.x * blockDim.x + threadIdx.x;
    if (n < N_NODES) deg[n] = 1.0f;  // self loop
}

__global__ void k_count_deg(const int* __restrict__ dst, float* __restrict__ deg) {
    int e = blockIdx.x * blockDim.x + threadIdx.x;
    if (e < N_EDGES) atomicAdd(&deg[dst[e]], 1.0f);
}

__global__ void k_finalize_deg(float* __restrict__ deg) {
    int n = blockIdx.x * blockDim.x + threadIdx.x;
    if (n < N_NODES) deg[n] = rsqrtf(deg[n]);  // deg >= 1 always
}

// ---------------- conversions ----------------
__global__ void k_f32_to_bf16(const float* __restrict__ in, __bf16* __restrict__ out, int n) {
    int i = blockIdx.x * blockDim.x + threadIdx.x;
    if (i < n) out[i] = (__bf16)in[i];
}

// in: rows x cols (row-major f32) -> out: cols x rows (row-major bf16)
__global__ void k_transpose_bf16(const float* __restrict__ in, __bf16* __restrict__ out,
                                 int rows, int cols) {
    int i = blockIdx.x * blockDim.x + threadIdx.x;
    if (i >= rows * cols) return;
    int c = i % cols, r = i / cols;
    out[(size_t)c * rows + r] = (__bf16)in[(size_t)r * cols + c];
}

__global__ void k_relu_to_bf16(const float* __restrict__ in, __bf16* __restrict__ out, int n) {
    int i = blockIdx.x * blockDim.x + threadIdx.x;
    if (i < n) {
        float v = in[i];
        out[i] = (__bf16)(v > 0.0f ? v : 0.0f);
    }
}

// ---------------- WMMA GEMM ----------------
// C[M,N] = A[M,K] (row-major bf16) * Bt[N,K] (B column-major, stored row-major bf16)
// one wave computes one 16x16 tile; K stepped by 32 per v_wmma_f32_16x16x32_bf16.
// A 16-bit layout (ISA 7.12.2): lane l, half=l>>4, m=l&15:
//   elems 0..7  -> K = k0 + half*8 + j
//   elems 8..15 -> K = k0 + 16 + half*8 + j
// B (column-major mirror): lane n&15 selects column, same K packing.
// C/D: lane l, vgpr r -> row = half*8 + r, col = l&15.
template <int M, int N, int K>
__global__ void k_wmma_gemm(const __bf16* __restrict__ A,
                            const __bf16* __restrict__ Bt,
                            float* __restrict__ C) {
    const int lane = threadIdx.x & 31;
    const int wave = threadIdx.x >> 5;
    const int tile = blockIdx.x * (blockDim.x >> 5) + wave;
    constexpr int tilesN = N >> 4;
    const int tm = tile / tilesN;
    const int tn = tile % tilesN;
    if (tm * 16 >= M) return;  // uniform per wave: EXEC stays all-ones for WMMA

    const int half = lane >> 4;
    const int idx  = lane & 15;
    const __bf16* arow = A  + (size_t)(tm * 16 + idx) * K;
    const __bf16* bcol = Bt + (size_t)(tn * 16 + idx) * K;

    v8f acc = {};
#pragma unroll
    for (int k0 = 0; k0 < K; k0 += 32) {
        v8bf alo = *(const v8bf*)(arow + k0 + half * 8);
        v8bf ahi = *(const v8bf*)(arow + k0 + 16 + half * 8);
        v8bf blo = *(const v8bf*)(bcol + k0 + half * 8);
        v8bf bhi = *(const v8bf*)(bcol + k0 + 16 + half * 8);
        v16bf av = __builtin_shufflevector(alo, ahi, 0,1,2,3,4,5,6,7,8,9,10,11,12,13,14,15);
        v16bf bv = __builtin_shufflevector(blo, bhi, 0,1,2,3,4,5,6,7,8,9,10,11,12,13,14,15);
        acc = __builtin_amdgcn_wmma_f32_16x16x32_bf16(
            /*neg_a=*/false, av, /*neg_b=*/false, bv,
            /*c_mod=*/(short)0, acc, /*reuse_a=*/false, /*reuse_b=*/false);
    }

    float* cptr = C + (size_t)(tm * 16 + half * 8) * N + tn * 16 + idx;
#pragma unroll
    for (int r = 0; r < 8; ++r) cptr[(size_t)r * N] = acc[r];
}

// ---------------- scatter (message passing) ----------------
// init: agg[n][c] = lin[n][c]*inv[n]^2 + bias[c]   (self loop + bias)
template <int C>
__global__ void k_self_bias(const float* __restrict__ lin, const float* __restrict__ inv,
                            const float* __restrict__ bias, float* __restrict__ agg) {
    int i = blockIdx.x * blockDim.x + threadIdx.x;
    if (i >= N_NODES * C) return;
    int n = i / C, c = i % C;
    float iv = inv[n];
    agg[i] = lin[i] * iv * iv + bias[c];
}

// edge-parallel: agg[dst][c] += lin[src][c] * inv[src]*inv[dst], 4 channels/thread
template <int C>
__global__ void k_edge_scatter(const float* __restrict__ lin,
                               const int* __restrict__ src, const int* __restrict__ dst,
                               const float* __restrict__ inv, float* __restrict__ agg) {
    constexpr int CH4 = C / 4;
    int t = blockIdx.x * blockDim.x + threadIdx.x;
    if (t >= N_EDGES * CH4) return;
    int e  = t / CH4;
    int c0 = (t % CH4) * 4;
    int s = src[e], d = dst[e];
    float coef = inv[s] * inv[d];
    const float* lrow = lin + (size_t)s * C + c0;
    float* arow = agg + (size_t)d * C + c0;
#pragma unroll
    for (int j = 0; j < 4; ++j) atomicAdd(&arow[j], lrow[j] * coef);
}

// ---------------- fused adjacency: out = sum_k w[k]*A_k ----------------
__global__ void k_fused_sum(const float* __restrict__ subg, const float* __restrict__ w,
                            float* __restrict__ out) {
    constexpr size_t TOTAL4 = (size_t)N_NODES * N_NODES / 4;
    size_t i = (size_t)blockIdx.x * blockDim.x + threadIdx.x;
    if (i >= TOTAL4) return;
    float w0 = w[0], w1 = w[1], w2 = w[2], w3 = w[3];
    const v4f* a0 = (const v4f*)subg;
    const v4f* a1 = a0 + TOTAL4;
    const v4f* a2 = a1 + TOTAL4;
    const v4f* a3 = a2 + TOTAL4;
    v4f r = __builtin_nontemporal_load(&a0[i]) * w0
          + __builtin_nontemporal_load(&a1[i]) * w1
          + __builtin_nontemporal_load(&a2[i]) * w2
          + __builtin_nontemporal_load(&a3[i]) * w3;
    __builtin_nontemporal_store(r, (v4f*)out + i);
}

// ---------------- launch ----------------
extern "C" void kernel_launch(void* const* d_in, const int* in_sizes, int n_in,
                              void* d_out, int out_size, void* d_ws, size_t ws_size,
                              hipStream_t stream) {
    const float* x    = (const float*)d_in[0];
    const int*   ei   = (const int*)  d_in[1];   // [2, E]: src then dst
    const float* subg = (const float*)d_in[2];
    // d_in[3] = node_types (unused by reference math)
    const float* W1   = (const float*)d_in[4];
    const float* b1   = (const float*)d_in[5];
    const float* W2   = (const float*)d_in[6];
    const float* b2   = (const float*)d_in[7];
    const float* subw = (const float*)d_in[8];

    const int* src = ei;
    const int* dst = ei + N_EDGES;

    char* ws = (char*)d_ws;
    float*  deg   = (float*) (ws + OFF_DEG);
    __bf16* xbf   = (__bf16*)(ws + OFF_XBF);
    __bf16* w1t   = (__bf16*)(ws + OFF_W1T);
    float*  hlin  = (float*) (ws + OFF_HLIN);
    float*  agg1  = (float*) (ws + OFF_AGG1);
    __bf16* hbf   = (__bf16*)(ws + OFF_HBF);
    __bf16* w2t   = (__bf16*)(ws + OFF_W2T);
    float*  h2lin = (float*) (ws + OFF_H2);

    float* emb   = (float*)d_out;                            // [N, OUT_C]
    float* fused = (float*)d_out + (size_t)N_NODES * OUT_C;  // [N, N]

    const int T = 256;

    // --- degrees ---
    hipLaunchKernelGGL(k_init_deg,     dim3(N_NODES / T), dim3(T), 0, stream, deg);
    hipLaunchKernelGGL(k_count_deg,    dim3(N_EDGES / T), dim3(T), 0, stream, dst, deg);
    hipLaunchKernelGGL(k_finalize_deg, dim3(N_NODES / T), dim3(T), 0, stream, deg);

    // --- bf16 operand prep for GEMM1 ---
    hipLaunchKernelGGL(k_f32_to_bf16,   dim3((N_NODES * IN_C) / T), dim3(T), 0, stream,
                       x, xbf, N_NODES * IN_C);
    hipLaunchKernelGGL(k_transpose_bf16, dim3((IN_C * HID_C) / T), dim3(T), 0, stream,
                       W1, w1t, IN_C, HID_C);

    // --- GEMM1: hlin = x @ W1  (4096x512x256) ---
    {
        constexpr int tiles = (N_NODES / 16) * (HID_C / 16);  // 4096
        hipLaunchKernelGGL((k_wmma_gemm<N_NODES, HID_C, IN_C>),
                           dim3(tiles / 4), dim3(128), 0, stream, xbf, w1t, hlin);
    }

    // --- conv1 aggregation: self+bias init, then edge atomics ---
    hipLaunchKernelGGL((k_self_bias<HID_C>), dim3((N_NODES * HID_C) / T), dim3(T), 0, stream,
                       hlin, deg, b1, agg1);
    hipLaunchKernelGGL((k_edge_scatter<HID_C>),
                       dim3((N_EDGES * (HID_C / 4)) / T), dim3(T), 0, stream,
                       hlin, src, dst, deg, agg1);

    // --- relu + bf16 for GEMM2 ---
    hipLaunchKernelGGL(k_relu_to_bf16, dim3((N_NODES * HID_C) / T), dim3(T), 0, stream,
                       agg1, hbf, N_NODES * HID_C);
    hipLaunchKernelGGL(k_transpose_bf16, dim3((HID_C * OUT_C) / T), dim3(T), 0, stream,
                       W2, w2t, HID_C, OUT_C);

    // --- GEMM2: h2lin = relu(h) @ W2  (4096x256x128) ---
    {
        constexpr int tiles = (N_NODES / 16) * (OUT_C / 16);  // 2048
        hipLaunchKernelGGL((k_wmma_gemm<N_NODES, OUT_C, HID_C>),
                           dim3(tiles / 4), dim3(128), 0, stream, hbf, w2t, h2lin);
    }

    // --- conv2 aggregation straight into d_out ---
    hipLaunchKernelGGL((k_self_bias<OUT_C>), dim3((N_NODES * OUT_C) / T), dim3(T), 0, stream,
                       h2lin, deg, b2, emb);
    hipLaunchKernelGGL((k_edge_scatter<OUT_C>),
                       dim3((N_EDGES * (OUT_C / 4)) / T), dim3(T), 0, stream,
                       h2lin, src, dst, deg, emb);

    // --- fused adjacency (streaming, NT) ---
    {
        constexpr size_t total4 = (size_t)N_NODES * N_NODES / 4;
        hipLaunchKernelGGL(k_fused_sum, dim3((unsigned)(total4 / T)), dim3(T), 0, stream,
                           subg, subw, fused);
    }
}
